// SelfAttention_29420525978199
// MI455X (gfx1250) — matrix operand order repaired and checked by reading, not compiled, not run on hardware
//
#include <hip/hip_runtime.h>
#include <hip/hip_bf16.h>
#include <math.h>

// ---- CDNA5 WMMA types ----
typedef __attribute__((ext_vector_type(16))) __bf16 v16bf;
typedef __attribute__((ext_vector_type(8)))  float  v8f;
typedef __attribute__((ext_vector_type(4)))  unsigned int ui4;
typedef __attribute__((ext_vector_type(2)))  unsigned int ui2;
typedef __attribute__((ext_vector_type(4)))  float f4;

constexpr int DIM = 64;     // D
constexpr int SEQ = 60;     // S
constexpr int NPART = 40;   // P
constexpr int NBATCH = 256; // B
constexpr int LDL = 72;     // padded LDS stride (u16): 144B rows -> conflict-free b128 frags

union Frag { v16bf v; ui4 q[2]; };

// f32 -> bf16 round-to-nearest-even (staging only)
__device__ __forceinline__ unsigned short f2bf(float f) {
    unsigned u = __builtin_bit_cast(unsigned, f);
    u = (u + 0x7FFFu + ((u >> 16) & 1u)) >> 16;
    return (unsigned short)u;
}
__device__ __forceinline__ unsigned int f2bf2(float a, float b) {
    return (unsigned int)f2bf(a) | ((unsigned int)f2bf(b) << 16);
}
// f32 -> bf16 truncation (hot-path stores; 1 VALU op, may select ds_store_b16_d16_hi)
__device__ __forceinline__ unsigned short f2bf_trunc(float f) {
    return (unsigned short)(__builtin_bit_cast(unsigned, f) >> 16);
}

// A fragment (16x32 bf16) from row-major [M][K] LDS (stride LDL), k-origin k0.
// Lane m=lane&15, hi=lane>>4: V0-3 hold K=k0+hi*8+{0..7}, V4-7 hold K=k0+16+hi*8+{0..7}.
__device__ __forceinline__ void loadA(const unsigned short* base, int k0, int lane, Frag& a) {
    const int m = lane & 15, hi = lane >> 4;
    const ui4* rp = (const ui4*)(base + m * LDL + k0);
    a.q[0] = rp[hi];
    a.q[1] = rp[2 + hi];
}

// B fragment (32x16 bf16) from a COLUMN-MAJOR [N][K] LDS matrix (stride LDL).
__device__ __forceinline__ void loadB(const unsigned short* base, int k0, int n0, int lane, Frag& b) {
    const int n  = n0 + (lane & 15);
    const int kb = k0;
    const ui4* p = (const ui4*)(base + n * LDL + kb);
    b.q[0] = p[(lane >> 4)];
    b.q[1] = p[2 + (lane >> 4)];
}

// C[4] (16x64 strip) = A(16x64, row-major LDS) x B(64x64, column-major LDS).
// Accumulator starts at constant zero -> inline-0 SRC2 on the first WMMA.
__device__ __forceinline__ void mm_strip(const unsigned short* A0, const unsigned short* B0,
                                         int lane, v8f c[4]) {
    Frag a0, a1;
    loadA(A0, 0,  lane, a0);
    loadA(A0, 32, lane, a1);
#pragma unroll
    for (int t = 0; t < 4; ++t) {
        Frag b0, b1;
        loadB(B0, 0,  t * 16, lane, b0);
        loadB(B0, 32, t * 16, lane, b1);
        v8f acc = {0.f, 0.f, 0.f, 0.f, 0.f, 0.f, 0.f, 0.f};
        acc = __builtin_amdgcn_wmma_f32_16x16x32_bf16(false, a0.v, false, b0.v,
                                                      (short)0, acc, false, false);
        acc = __builtin_amdgcn_wmma_f32_16x16x32_bf16(false, a1.v, false, b1.v,
                                                      (short)0, acc, false, false);
        c[t] = acc;
    }
}

// C store (+optional bias, +optional relu) to row-major bf16 strip (stride LDL).
// bias/relu are compile-time-foldable at each inlined call site.
__device__ __forceinline__ void storeC(unsigned short* dst, const v8f c[4],
                                       const float* bias, bool relu, int lane) {
    const int n = lane & 15, hi = lane >> 4;
#pragma unroll
    for (int t = 0; t < 4; ++t) {
        const float bv = bias ? bias[t * 16 + n] : 0.0f;
#pragma unroll
        for (int r = 0; r < 8; ++r) {
            float v = c[t][r];
            if (bias) v += bv;
            if (relu) v = fmaxf(v, 0.0f);
            dst[(r + hi * 8) * LDL + t * 16 + n] = f2bf_trunc(v);
        }
    }
}

// Stage a 64x64 f32 global matrix into LDS bf16 (same element order), float4 reads,
// 64-bit packed LDS writes.  dst stride LDL.
__device__ __forceinline__ void stageW(const float* __restrict__ src, unsigned short* dst, int tid) {
    const f4* sv = (const f4*)src;
    ui2* dv = (ui2*)dst;
    for (int g = tid; g < (DIM * DIM) / 4; g += 128) {
        const int e = g >> 4, dq = g & 15;
        f4 f = sv[g];
        ui2 p; p[0] = f2bf2(f[0], f[1]); p[1] = f2bf2(f[2], f[3]);
        dv[(LDL / 4) * e + dq] = p;
    }
}

__global__ __launch_bounds__(128)
void attn_head_kernel(const float* __restrict__ x,
                      const float* __restrict__ Wq, const float* __restrict__ bq,
                      const float* __restrict__ Wk, const float* __restrict__ bk,
                      const float* __restrict__ Wv, const float* __restrict__ bv,
                      const float* __restrict__ W1, const float* __restrict__ b1,
                      const float* __restrict__ W2, const float* __restrict__ b2,
                      const float* __restrict__ W3, const float* __restrict__ b3,
                      float* __restrict__ out)
{
    // Row-major A operands / per-wave scratch strips:
    __shared__ __align__(16) unsigned short xs[DIM * LDL];     // x tile; later Q/attn/h strips
    __shared__ __align__(16) unsigned short Ks[DIM * LDL];     // K row-major == B col-major for scores
    // Column-major B operands:
    __shared__ __align__(16) unsigned short slot0[DIM * LDL];  // Wq (later W1)
    __shared__ __align__(16) unsigned short slot1[DIM * LDL];  // Wk (later W2)
    __shared__ __align__(16) unsigned short slot2[DIM * LDL];  // Wv
    __shared__ __align__(16) unsigned short Vt[DIM * LDL];     // V transposed: Vt[n=d][k=keyrow]
    __shared__ float sb[6 * 64];                               // bq bk bv b1 b2 w3

    const int tid  = threadIdx.x;
    const int lane = tid & 31;
    const int wave = tid >> 5;
    const int blk  = blockIdx.x;                 // b*NPART + p
    const int n = lane & 15, hi = lane >> 4;

    // ---- Phase A1: stage x + QKV weights + biases ----
    {
        const f4* xv = (const f4*)(x + (size_t)blk * SEQ * DIM);
        ui2* dv = (ui2*)xs;
        for (int g = tid; g < (SEQ * DIM) / 4; g += 128) {
            const int s = g >> 4, dq = g & 15;
            f4 f = xv[g];
            ui2 p; p[0] = f2bf2(f[0], f[1]); p[1] = f2bf2(f[2], f[3]);
            dv[(LDL / 4) * s + dq] = p;
        }
        for (int i = tid; i < 4 * 16; i += 128) {       // zero rows 60..63
            const int r = SEQ + (i >> 4), dq = i & 15;
            ui2 z; z[0] = 0u; z[1] = 0u;
            dv[(LDL / 4) * r + dq] = z;
        }
    }
    stageW(Wq, slot0, tid);
    stageW(Wk, slot1, tid);
    stageW(Wv, slot2, tid);
    if (tid < 64) {
        sb[tid]       = bq[tid];
        sb[64 + tid]  = bk[tid];
        sb[128 + tid] = bv[tid];
        sb[192 + tid] = b1[tid];
        sb[256 + tid] = b2[tid];
        sb[320 + tid] = W3[tid];
    }
    __syncthreads();

    const int row0 = wave * 16;
    unsigned short* strip = xs + row0 * LDL;     // wave-private strip (x rows, then Q/attn/h)
    v8f c[4];

    // ---- Phase B: K -> Ks (row-major), V -> Vt (transposed), Q last -> own xs strip ----
    mm_strip(strip, slot1, lane, c);                     // K
    storeC(Ks + row0 * LDL, c, sb + 64, false, lane);
    mm_strip(strip, slot2, lane, c);                     // V
    {
        const float* bvv = sb + 128;
#pragma unroll
        for (int t = 0; t < 4; ++t) {
            const float bv2 = bvv[t * 16 + n];
#pragma unroll
            for (int r = 0; r < 8; ++r)
                Vt[(t * 16 + n) * LDL + (row0 + r + hi * 8)] = f2bf_trunc(c[t][r] + bv2);
        }
    }
    mm_strip(strip, slot0, lane, c);                     // Q (last read of x strip)
    storeC(strip, c, sb, false, lane);                   // overwrite own strip with Q
    __syncthreads();                                     // Ks/Vt complete; QKV weights dead

    // ---- Phase A2: recycle slot0/1 with W1/W2 (consumed after next barrier) ----
    stageW(W1, slot0, tid);
    stageW(W2, slot1, tid);

    // ---- Phase C: scores = Q K^T / sqrt(D); softmax over 60 valid cols ----
    mm_strip(strip, Ks, lane, c);
    {
        const float scale = 0.125f;                      // 1/sqrt(64)
        float ex[4][8];
#pragma unroll
        for (int r = 0; r < 8; ++r) {
            float mx = -INFINITY;
#pragma unroll
            for (int t = 0; t < 4; ++t) {
                const int col = t * 16 + n;
                const float v = (col < SEQ) ? c[t][r] * scale : -INFINITY;
                ex[t][r] = v;
                mx = fmaxf(mx, v);
            }
#pragma unroll
            for (int off = 1; off < 16; off <<= 1) mx = fmaxf(mx, __shfl_xor(mx, off, 16));
            float sum = 0.0f;
#pragma unroll
            for (int t = 0; t < 4; ++t) {
                const int col = t * 16 + n;
                const float e = (col < SEQ) ? __expf(ex[t][r] - mx) : 0.0f;
                ex[t][r] = e;
                sum += e;
            }
#pragma unroll
            for (int off = 1; off < 16; off <<= 1) sum += __shfl_xor(sum, off, 16);
            const float inv = 1.0f / sum;
#pragma unroll
            for (int t = 0; t < 4; ++t)
                strip[(r + hi * 8) * LDL + t * 16 + n] = f2bf_trunc(ex[t][r] * inv);
        }
    }

    // ---- Phase D: weighted = attn x V ----
    mm_strip(strip, Vt, lane, c);
    storeC(strip, c, nullptr, false, lane);
    __syncthreads();                                     // W1/W2 staged by all threads

    // ---- Phase E/F: two ReLU MLP layers ----
    mm_strip(strip, slot0, lane, c);
    storeC(strip, c, sb + 192, true, lane);
    mm_strip(strip, slot1, lane, c);

    // ---- Phase G: out[s] = relu(h2[s,:] + b2) . W3 + b3 ----
    float acc[8];
#pragma unroll
    for (int r = 0; r < 8; ++r) {
        float a = 0.0f;
#pragma unroll
        for (int t = 0; t < 4; ++t)
            a += fmaxf(c[t][r] + sb[256 + t * 16 + n], 0.0f) * sb[320 + t * 16 + n];
#pragma unroll
        for (int off = 1; off < 16; off <<= 1) a += __shfl_xor(a, off, 16);
        acc[r] = a;
    }
    if (n == 0) {
        const float bb = b3[0];
#pragma unroll
        for (int r = 0; r < 8; ++r) {
            const int s = row0 + r + hi * 8;
            if (s < SEQ) out[(size_t)blk * SEQ + s] = acc[r] + bb;
        }
    }
}

extern "C" void kernel_launch(void* const* d_in, const int* in_sizes, int n_in,
                              void* d_out, int out_size, void* d_ws, size_t ws_size,
                              hipStream_t stream) {
    const float* x  = (const float*)d_in[0];
    const float* Wq = (const float*)d_in[1];
    const float* bq = (const float*)d_in[2];
    const float* Wk = (const float*)d_in[3];
    const float* bk = (const float*)d_in[4];
    const float* Wv = (const float*)d_in[5];
    const float* bv = (const float*)d_in[6];
    const float* W1 = (const float*)d_in[7];
    const float* b1 = (const float*)d_in[8];
    const float* W2 = (const float*)d_in[9];
    const float* b2 = (const float*)d_in[10];
    const float* W3 = (const float*)d_in[11];
    const float* b3 = (const float*)d_in[12];
    float* out = (float*)d_out;

    attn_head_kernel<<<NBATCH * NPART, 128, 0, stream>>>(
        x, Wq, bq, Wk, bk, Wv, bv, W1, b1, W2, b2, W3, b3, out);
}
